// NonLocalAttention_12017318494239
// MI455X (gfx1250) — matrix operand order repaired and checked by reading, not compiled, not run on hardware
//
#include <hip/hip_runtime.h>

typedef __attribute__((ext_vector_type(16))) _Float16 v16h;
typedef __attribute__((ext_vector_type(8)))  _Float16 h8;
typedef __attribute__((ext_vector_type(8)))  float    v8f;

#define HW 6400
#define C  64
#define NB 4          // batches
#define NEG_SLOPE 0.2f

static __device__ __forceinline__ v8f wmma_f16(v16h a, v16h b, v8f c) {
  // D = A(16x32 f16) * B(32x16 f16) + C(16x16 f32)
  return __builtin_amdgcn_wmma_f32_16x16x32_f16(false, a, false, b,
                                                (short)0, c, false, false);
}

// ---------------------------------------------------------------------------
// Kernel 1: e_i = leaky_relu(w_i @ x + b_i) for i=1..3, emitted as f16.
//   q (e1): [n][HW][C]   (hw-major, c contiguous)  -> A-fragments for S
//   k (e2): [n][HW][C]   (hw-major, c contiguous)  -> B-fragments for S
//   v (e3): [n][C][HW]   (c-major, hw contiguous)  -> B-fragments for P@V
// One wave handles one (batch, 16-wide hw tile), all 64 output rows, all 3 w.
// ---------------------------------------------------------------------------
__global__ __launch_bounds__(128) void qkv_kernel(
    const float* __restrict__ x,
    const float* __restrict__ w1, const float* __restrict__ b1,
    const float* __restrict__ w2, const float* __restrict__ b2,
    const float* __restrict__ w3, const float* __restrict__ b3,
    _Float16* __restrict__ qb, _Float16* __restrict__ kb,
    _Float16* __restrict__ vb)
{
  const int lane = threadIdx.x & 31;
  const int wv   = threadIdx.x >> 5;
  const int gw   = blockIdx.x * 4 + wv;
  const int batch  = gw / (HW / 16);
  const int hwt    = gw % (HW / 16);
  const int hwBase = hwt * 16;
  const int half = lane >> 4;   // which 16-lane half of the wave
  const int l16  = lane & 15;

  const float* xb = x + (size_t)batch * C * HW;
  const float* W[3] = {w1, w2, w3};
  const float* B[3] = {b1, b2, b3};

  v8f acc[3][4];
  #pragma unroll
  for (int wi = 0; wi < 3; ++wi)
    #pragma unroll
    for (int ot = 0; ot < 4; ++ot) acc[wi][ot] = (v8f){};

  const int hw = hwBase + l16;

  for (int ck = 0; ck < C; ck += 32) {
    // B fragment of x: B[K=c][N=hw]; lane holds col hw, halfs -> c
    v16h bx;
    #pragma unroll
    for (int h = 0; h < 16; ++h) {
      const int cc = ck + half * 8 + (h & 7) + (h >> 3) * 16;
      bx[h] = (_Float16)xb[cc * HW + hw];
    }
    #pragma unroll
    for (int wi = 0; wi < 3; ++wi) {
      #pragma unroll
      for (int ot = 0; ot < 4; ++ot) {
        // A fragment of w: lane holds row o; halfs 0-7 -> c=kb..kb+7,
        // halfs 8-15 -> c=kb+16..kb+23, kb = 8*half
        const int o = ot * 16 + l16;
        const float* wr = W[wi] + o * C + ck + half * 8;
        v16h a;
        #pragma unroll
        for (int j = 0; j < 8; ++j) {
          a[j]     = (_Float16)wr[j];
          a[8 + j] = (_Float16)wr[16 + j];
        }
        acc[wi][ot] = wmma_f16(a, bx, acc[wi][ot]);
      }
    }
  }

  // Epilogue: bias + leaky relu + f16 store.
  // D layout: lane -> col hw = hwBase+l16 ; reg r -> row o = ot*16 + r + 8*half
  #pragma unroll
  for (int wi = 0; wi < 3; ++wi) {
    #pragma unroll
    for (int ot = 0; ot < 4; ++ot) {
      h8 pk;
      #pragma unroll
      for (int r = 0; r < 8; ++r) {
        const int o = ot * 16 + r + 8 * half;
        float val = acc[wi][ot][r] + B[wi][o];
        val = (val >= 0.0f) ? val : NEG_SLOPE * val;
        pk[r] = (_Float16)val;
      }
      if (wi == 2) {
        // v: [c][hw], strided over o
        #pragma unroll
        for (int r = 0; r < 8; ++r) {
          const int o = ot * 16 + r + 8 * half;
          vb[(size_t)batch * C * HW + (size_t)o * HW + hw] = pk[r];
        }
      } else {
        // q/k: [hw][c]; consecutive regs = consecutive o -> one 16B store
        _Float16* dst = (wi == 0 ? qb : kb) +
                        (size_t)batch * HW * C + (size_t)hw * C +
                        ot * 16 + 8 * half;
        *(h8*)dst = pk;
      }
    }
  }
}

// ---------------------------------------------------------------------------
// Kernel 2: flash attention, 64 keys per iteration (100 iterations).
// One wave per (batch, 16-query tile).
//   S 16x64: 8 WMMAs  -> online softmax (one shfl-max chain + one shfl-sum
//   chain + one rescale per 64 keys) -> P via LDS (D->A transpose)
//   -> O += P@V: 8 WMMAs.
// ---------------------------------------------------------------------------
__global__ __launch_bounds__(128) void attn_kernel(
    const _Float16* __restrict__ qb,   // [n][HW][C]
    const _Float16* __restrict__ kbf,  // [n][HW][C]
    const _Float16* __restrict__ vbf,  // [n][C][HW]
    float* __restrict__ out)           // [n][C][HW]
{
  __shared__ _Float16 pshare[4][16][64];   // per-wave P tile slab (8 KB)

  const int lane = threadIdx.x & 31;
  const int wv   = threadIdx.x >> 5;
  const int gw   = blockIdx.x * 4 + wv;
  const int batch = gw / (HW / 16);
  const int qt    = gw % (HW / 16);
  const int qBase = qt * 16;
  const int half = lane >> 4;
  const int l16  = lane & 15;

  const _Float16* qp = qb  + (size_t)batch * HW * C;
  const _Float16* kp = kbf + (size_t)batch * HW * C;
  const _Float16* vp = vbf + (size_t)batch * C * HW;

  // Q A-fragments (held for whole loop): lane -> query row qBase+l16
  v16h qa[2];
  #pragma unroll
  for (int ci = 0; ci < 2; ++ci) {
    const _Float16* src = qp + (size_t)(qBase + l16) * C + ci * 32 + half * 16;
    h8 lo = *(const h8*)(src);
    h8 hi = *(const h8*)(src + 8);
    v16h a;
    #pragma unroll
    for (int j = 0; j < 8; ++j) { a[j] = lo[j]; a[8 + j] = hi[j]; }
    qa[ci] = a;
  }

  v8f o_acc[4];
  #pragma unroll
  for (int ct = 0; ct < 4; ++ct) o_acc[ct] = (v8f){};
  float m_run[8], l_run[8];
  #pragma unroll
  for (int r = 0; r < 8; ++r) { m_run[r] = -3.0e38f; l_run[r] = 0.0f; }

  for (int kc = 0; kc < HW; kc += 64) {
    // ---- S tile 16x64: four 16-key column tiles, K=c=64 (2 WMMAs each)
    v8f s[4];
    #pragma unroll
    for (int nt = 0; nt < 4; ++nt) {
      const int key = kc + nt * 16 + l16;   // lane -> key column
      v8f a = (v8f){};
      #pragma unroll
      for (int ci = 0; ci < 2; ++ci) {
        v16h bf = *(const v16h*)(kp + (size_t)key * C + ci * 32 + half * 16);
        a = wmma_f16(qa[ci], bf, a);
      }
      s[nt] = a;
    }

    // ---- Online softmax. Row of reg r is (r + 8*half); its 16 cols live
    // across lanes 0-15 (half 0) / 16-31 (half 1) -> width-16 reductions.
    #pragma unroll
    for (int r = 0; r < 8; ++r) {
      float mx = fmaxf(fmaxf(s[0][r], s[1][r]), fmaxf(s[2][r], s[3][r]));
      #pragma unroll
      for (int off = 8; off >= 1; off >>= 1)
        mx = fmaxf(mx, __shfl_xor(mx, off, 16));
      const float m_new = fmaxf(m_run[r], mx);
      const float corr  = __expf(m_run[r] - m_new);
      float p[4], rs = 0.0f;
      #pragma unroll
      for (int nt = 0; nt < 4; ++nt) {
        p[nt] = __expf(s[nt][r] - m_new);
        rs += p[nt];
      }
      #pragma unroll
      for (int off = 8; off >= 1; off >>= 1)
        rs += __shfl_xor(rs, off, 16);
      l_run[r] = l_run[r] * corr + rs;
      m_run[r] = m_new;
      #pragma unroll
      for (int ct = 0; ct < 4; ++ct) o_acc[ct][r] *= corr;
      // stash P in LDS row-major [query][key]
      #pragma unroll
      for (int nt = 0; nt < 4; ++nt)
        pshare[wv][r + 8 * half][l16 + nt * 16] = (_Float16)p[nt];
    }
    asm volatile("s_wait_dscnt 0" ::: "memory");
    __builtin_amdgcn_wave_barrier();

    // ---- Re-read P as two A fragments (key sub-chunks 0-31 and 32-63):
    // lane -> row l16; halfs 0-15 K=kb..kb+15, kb = 16*half
    v16h pa[2];
    #pragma unroll
    for (int kk = 0; kk < 2; ++kk) {
      const _Float16* src = &pshare[wv][l16][kk * 32 + half * 16];
      v16h a;
      #pragma unroll
      for (int j = 0; j < 8; ++j) { a[j] = src[j]; a[8 + j] = src[8 + j]; }
      pa[kk] = a;
    }
    __builtin_amdgcn_wave_barrier();

    // ---- O += P @ V ; V B-fragment: lane -> c col, halfs -> contiguous keys
    #pragma unroll
    for (int ct = 0; ct < 4; ++ct) {
      const int c = ct * 16 + l16;
      #pragma unroll
      for (int kk = 0; kk < 2; ++kk) {
        v16h vf = *(const v16h*)(vp + (size_t)c * HW + kc + kk * 32 + half * 16);
        o_acc[ct] = wmma_f16(pa[kk], vf, o_acc[ct]);
      }
    }
  }

  // Normalize and store: out[n][c][hw]; consecutive regs r = consecutive
  // query rows -> stride-1 f32 stores.
  float* op = out + (size_t)batch * C * HW;
  #pragma unroll
  for (int ct = 0; ct < 4; ++ct) {
    const int c = ct * 16 + l16;
    const float inv = 1.0f;  // keep divide exact per-row below
    (void)inv;
    #pragma unroll
    for (int r = 0; r < 8; ++r) {
      const int qrow = qBase + r + 8 * half;
      op[(size_t)c * HW + qrow] = o_acc[ct][r] / l_run[r];
    }
  }
}

// ---------------------------------------------------------------------------
extern "C" void kernel_launch(void* const* d_in, const int* in_sizes, int n_in,
                              void* d_out, int out_size, void* d_ws, size_t ws_size,
                              hipStream_t stream) {
  (void)in_sizes; (void)n_in; (void)out_size; (void)ws_size;
  const float* x  = (const float*)d_in[0];
  const float* w1 = (const float*)d_in[1];
  const float* b1 = (const float*)d_in[2];
  const float* w2 = (const float*)d_in[3];
  const float* b2 = (const float*)d_in[4];
  const float* w3 = (const float*)d_in[5];
  const float* b3 = (const float*)d_in[6];
  float* out = (float*)d_out;

  _Float16* ws = (_Float16*)d_ws;
  const size_t per = (size_t)NB * HW * C;   // halfs per tensor
  _Float16* qb = ws;
  _Float16* kb = ws + per;
  _Float16* vb = ws + 2 * per;

  const int waves  = NB * (HW / 16);        // 1600 waves
  const int blocks = waves / 4;             // 4 waves (128 thr) per block

  qkv_kernel<<<blocks, 128, 0, stream>>>(x, w1, b1, w2, b2, w3, b3, qb, kb, vb);
  attn_kernel<<<blocks, 128, 0, stream>>>(qb, kb, vb, out);
}
